// TransformerEncoderLayer_59433757442290
// MI455X (gfx1250) — compile-verified, matrix operand order checked by reading, not run on hardware
//
#include <hip/hip_runtime.h>
#include <hip/hip_bf16.h>
#include <math.h>
#include <stdint.h>

// ---------------------------------------------------------------------------
// CDNA5 (gfx1250) transformer-encoder-layer pipeline.
// wave32; GEMMs: v_wmma_f32_16x16x32_f16, f16 operands staged to LDS with
// async global->LDS copies (ASYNCcnt), double-buffered; f32 accumulate.
// ---------------------------------------------------------------------------

typedef __attribute__((ext_vector_type(16))) _Float16 v16h;
typedef __attribute__((ext_vector_type(8)))  float    v8f;

#define BM 128
#define BN 64
#define BK 64
#define LDH 72   // padded LDS row stride (halves); 144B = 9*16 -> 16B aligned rows

// -------------------------------- utilities --------------------------------

__device__ __forceinline__ float warpReduceSum(float v) {
#pragma unroll
  for (int off = 16; off > 0; off >>= 1) v += __shfl_xor(v, off, 32);
  return v;
}

__device__ __forceinline__ void atomicMaxF(float* addr, float val) {
  unsigned* ua = (unsigned*)addr;
  unsigned old = *ua;
  while (__uint_as_float(old) < val) {
    unsigned assumed = old;
    old = atomicCAS(ua, assumed, __float_as_uint(val));
    if (old == assumed) break;
  }
}

__global__ void fill_kernel(float* __restrict__ p, float v, long long n) {
  long long i = (long long)blockIdx.x * 256 + threadIdx.x;
  if (i < n) p[i] = v;
}

__global__ void f32to16_kernel(const float* __restrict__ in, _Float16* __restrict__ out,
                               long long n) {
  long long i = (long long)blockIdx.x * 256 + threadIdx.x;
  if (i < n) out[i] = (_Float16)in[i];
}

// ------------------------------- LayerNorm ---------------------------------
// f32 in -> f16 out (output always feeds a WMMA GEMM)

__global__ __launch_bounds__(256)
void ln_kernel(const float* __restrict__ x, const float* __restrict__ g,
               const float* __restrict__ b, _Float16* __restrict__ y, int D) {
  __shared__ float sdata[256];
  int row = blockIdx.x, tid = threadIdx.x;
  x += (long long)row * D;
  y += (long long)row * D;
  float s = 0.f;
  for (int i = tid; i < D; i += 256) s += x[i];
  sdata[tid] = s; __syncthreads();
  for (int st = 128; st > 0; st >>= 1) { if (tid < st) sdata[tid] += sdata[tid + st]; __syncthreads(); }
  float mean = sdata[0] / (float)D;
  __syncthreads();
  float vs = 0.f;
  for (int i = tid; i < D; i += 256) { float d = x[i] - mean; vs += d * d; }
  sdata[tid] = vs; __syncthreads();
  for (int st = 128; st > 0; st >>= 1) { if (tid < st) sdata[tid] += sdata[tid + st]; __syncthreads(); }
  float rstd = rsqrtf(sdata[0] / (float)D + 1e-6f);
  for (int i = tid; i < D; i += 256) y[i] = (_Float16)((x[i] - mean) * rstd * g[i] + b[i]);
}

// --------------------------- WMMA GEMM (f16 in) ----------------------------
// C = epilogue( A[M,K] @ op(B)[K,N] ); A,B f16 in memory; f32 accumulate.
// bTrans=1: B row-major [N,K] (A @ B^T) -> async-copied straight to LDS.
// bTrans=0: B row-major [K,N]          -> transposed while staging (packed ds).
// Per block: 128x64 C tile, 8 waves, each wave a 32x32 quad (2x2 fragments)
//   -> 8 v_wmma_f32_16x16x32_f16 per wave per K-step, double-buffered LDS.

__global__ __launch_bounds__(256)
void gemm_wmma_f16(const _Float16* __restrict__ A, long long sA1, long long sA2, int bdivA, int lda,
                   const _Float16* __restrict__ Bm, long long sB1, long long sB2, int bdivB, int ldb,
                   int bTrans,
                   const float* __restrict__ bias,
                   const float* __restrict__ resid, int ldr,
                   float* __restrict__ Cf, _Float16* __restrict__ Ch,
                   long long sC1, long long sC2, int bdivC, int ldc,
                   int M, int Nmat, int K, float alpha, int relu) {
  int bz = blockIdx.z;
  const _Float16* Ab = A  + (long long)(bz / bdivA) * sA1 + (long long)(bz % bdivA) * sA2;
  const _Float16* Bb = Bm + (long long)(bz / bdivB) * sB1 + (long long)(bz % bdivB) * sB2;
  long long coff = (long long)(bz / bdivC) * sC1 + (long long)(bz % bdivC) * sC2;

  __shared__ __align__(16) _Float16 As[2][BM][LDH];
  __shared__ __align__(16) _Float16 Bs[2][BN][LDH];

  int tid = threadIdx.x, lane = tid & 31, wave = tid >> 5;
  int wm = (wave & 3) << 5;     // 0,32,64,96
  int wn = (wave >> 2) << 5;    // 0,32
  int m0 = blockIdx.y * BM;
  int n0 = blockIdx.x * BN;

  // staging maps (all problem dims are multiples of BM/BN/BK)
  int arow = tid >> 1, acol = (tid & 1) << 5;   // A: 128x64, 32 halves (64B)/thread
  int brow = tid >> 2, bcol = (tid & 3) << 4;   // B(trans): 64x64, 16 halves/thread
  int kk = (tid >> 3) << 1, nb = (tid & 7) << 3;// B(no-trans): 2k x 8n per thread

  // WMMA 16-bit fragment addressing (ISA 7.12.2, wave32)
  int row = lane & 15;
  int kb = (lane >> 4) << 3;

  v8f acc00 = {}, acc01 = {}, acc10 = {}, acc11 = {};

  auto stage = [&](int buf, int ks) {
    {  // A tile: 4 async b128 per thread (64B contiguous in LDS and global)
      unsigned la = (unsigned)(uintptr_t)&As[buf][arow][acol];
      const _Float16* ga = Ab + (long long)(m0 + arow) * lda + ks + acol;
      asm volatile("global_load_async_to_lds_b128 %0, %1, off"           :: "v"(la), "v"(ga) : "memory");
      asm volatile("global_load_async_to_lds_b128 %0, %1, off offset:16" :: "v"(la), "v"(ga) : "memory");
      asm volatile("global_load_async_to_lds_b128 %0, %1, off offset:32" :: "v"(la), "v"(ga) : "memory");
      asm volatile("global_load_async_to_lds_b128 %0, %1, off offset:48" :: "v"(la), "v"(ga) : "memory");
    }
    if (bTrans) {  // weight [N,K]: same K-contiguous layout, async copy
      unsigned lb = (unsigned)(uintptr_t)&Bs[buf][brow][bcol];
      const _Float16* gb = Bb + (long long)(n0 + brow) * ldb + ks + bcol;
      asm volatile("global_load_async_to_lds_b128 %0, %1, off"           :: "v"(lb), "v"(gb) : "memory");
      asm volatile("global_load_async_to_lds_b128 %0, %1, off offset:16" :: "v"(lb), "v"(gb) : "memory");
    } else {       // [K,N]: transpose while staging, pack (k,k+1) pairs
      const _Float16* g0 = Bb + (long long)(ks + kk) * ldb + n0 + nb;
      const _Float16* g1 = g0 + ldb;
#pragma unroll
      for (int j = 0; j < 8; ++j) {
        union { _Float16 h[2]; unsigned u; } p;
        p.h[0] = g0[j]; p.h[1] = g1[j];
        *(unsigned*)&Bs[buf][nb + j][kk] = p.u;
      }
    }
  };

  auto frag = [&](const _Float16 (*T)[LDH], int r0, int kofs) -> v16h {
    union { v16h v; unsigned u[8]; } t;
#pragma unroll
    for (int i = 0; i < 8; ++i) {
      int c = kofs + kb + 2 * i + ((i & 4) ? 8 : 0);
      t.u[i] = *(const unsigned*)&T[r0 + row][c];
    }
    return t.v;
  };

  stage(0, 0);
  asm volatile("s_wait_asynccnt 0x0" ::: "memory");
  __syncthreads();

  int buf = 0;
  for (int ks = 0; ks < K; ks += BK) {
    if (ks + BK < K) stage(buf ^ 1, ks + BK);   // overlap next-tile DMA with math
#pragma unroll
    for (int ksub = 0; ksub < 2; ++ksub) {
      int kofs = ksub << 5;
      v16h a0 = frag(As[buf], wm,      kofs);
      v16h a1 = frag(As[buf], wm + 16, kofs);
      v16h b0 = frag(Bs[buf], wn,      kofs);
      v16h b1 = frag(Bs[buf], wn + 16, kofs);
      acc00 = __builtin_amdgcn_wmma_f32_16x16x32_f16(false, a0, false, b0, (short)0, acc00, false, false);
      acc01 = __builtin_amdgcn_wmma_f32_16x16x32_f16(false, a0, false, b1, (short)0, acc01, false, false);
      acc10 = __builtin_amdgcn_wmma_f32_16x16x32_f16(false, a1, false, b0, (short)0, acc10, false, false);
      acc11 = __builtin_amdgcn_wmma_f32_16x16x32_f16(false, a1, false, b1, (short)0, acc11, false, false);
    }
    asm volatile("s_wait_asynccnt 0x0" ::: "memory");
    __syncthreads();
    buf ^= 1;
  }

  // C/D layout: VGPR v -> M = v + (lane>=16 ? 8 : 0), N = lane&15
  int gm0 = m0 + wm + ((lane >> 4) << 3);
  int cn0 = n0 + wn + (lane & 15);
  auto storeTile = [&](v8f a, int gmB, int cn) {
#pragma unroll
    for (int vv = 0; vv < 8; ++vv) {
      int gm = gmB + vv;
      float val = a[vv];
      if (bias) val += bias[cn];
      val *= alpha;
      if (relu) val = fmaxf(val, 0.f);
      if (resid) val += resid[(long long)gm * ldr + cn];
      if (Ch) Ch[coff + (long long)gm * ldc + cn] = (_Float16)val;
      else    Cf[coff + (long long)gm * ldc + cn] = val;
    }
  };
  storeTile(acc00, gm0,      cn0);
  storeTile(acc01, gm0,      cn0 + 16);
  storeTile(acc10, gm0 + 16, cn0);
  storeTile(acc11, gm0 + 16, cn0 + 16);
}

// --------------------------- GAT attention pieces --------------------------

__global__ __launch_bounds__(256)
void gat_att_kernel(const _Float16* __restrict__ h, const float* __restrict__ att_src,
                    const float* __restrict__ att_dst, float* __restrict__ asrc,
                    float* __restrict__ adst) {
  int gw = blockIdx.x * 8 + (threadIdx.x >> 5);  // over N*H
  int lane = threadIdx.x & 31;
  int n = gw >> 3, hh = gw & 7;
  const _Float16* hp = h + (long long)n * 4096 + hh * 512;
  const float* sa = att_src + hh * 512;
  const float* da = att_dst + hh * 512;
  float s = 0.f, d = 0.f;
  for (int c = lane; c < 512; c += 32) { float hv = (float)hp[c]; s += hv * sa[c]; d += hv * da[c]; }
  s = warpReduceSum(s); d = warpReduceSum(d);
  if (lane == 0) { asrc[gw] = s; adst[gw] = d; }
}

__device__ __forceinline__ float leaky02(float v) { return v > 0.f ? v : 0.2f * v; }

__global__ void edge_max_kernel(const int* __restrict__ ei, const float* __restrict__ asrc,
                                const float* __restrict__ adst, float* __restrict__ mbuf,
                                int E, int N) {
  int idx = blockIdx.x * 256 + threadIdx.x;
  if (idx >= (E + N) * 8) return;
  int e = idx >> 3, hh = idx & 7, s, d;
  if (e < E) { s = ei[e]; d = ei[E + e]; } else { s = d = e - E; }
  atomicMaxF(&mbuf[d * 8 + hh], leaky02(asrc[s * 8 + hh] + adst[d * 8 + hh]));
}

__global__ void edge_sum_kernel(const int* __restrict__ ei, const float* __restrict__ asrc,
                                const float* __restrict__ adst, const float* __restrict__ mbuf,
                                float* __restrict__ dbuf, int E, int N) {
  int idx = blockIdx.x * 256 + threadIdx.x;
  if (idx >= (E + N) * 8) return;
  int e = idx >> 3, hh = idx & 7, s, d;
  if (e < E) { s = ei[e]; d = ei[E + e]; } else { s = d = e - E; }
  float v = leaky02(asrc[s * 8 + hh] + adst[d * 8 + hh]);
  atomicAdd(&dbuf[d * 8 + hh], __expf(v - mbuf[d * 8 + hh]));
}

__global__ __launch_bounds__(256)
void edge_agg_kernel(const int* __restrict__ ei, const float* __restrict__ asrc,
                     const float* __restrict__ adst, const float* __restrict__ mbuf,
                     const float* __restrict__ dbuf, const _Float16* __restrict__ h,
                     float* __restrict__ agg, int E, int N) {
  int e = blockIdx.x, s, d;
  if (e < E) { s = ei[e]; d = ei[E + e]; } else { s = d = e - E; }
  const _Float16* hs = h + (long long)s * 4096;
  float* ad = agg + (long long)d * 4096;
  for (int hh = 0; hh < 8; ++hh) {
    float v = leaky02(asrc[s * 8 + hh] + adst[d * 8 + hh]);
    float w = __expf(v - mbuf[d * 8 + hh]) / dbuf[d * 8 + hh];
    for (int c = threadIdx.x; c < 512; c += 256)
      atomicAdd(&ad[hh * 512 + c], w * (float)hs[hh * 512 + c]);
  }
}

__global__ void head_mean_kernel(const float* __restrict__ agg, const float* __restrict__ gbias,
                                 float* __restrict__ gnn, long long n) {
  long long i = (long long)blockIdx.x * 256 + threadIdx.x;
  if (i >= n) return;
  int nn = (int)(i >> 9), dd = (int)(i & 511);
  float s = 0.f;
#pragma unroll
  for (int hh = 0; hh < 8; ++hh) s += agg[(long long)nn * 4096 + hh * 512 + dd];
  gnn[i] = s * 0.125f + gbias[dd];
}

// ------------------------------- softmax -----------------------------------
// f32 scores in -> f16 attn out (feeds ctx GEMM)

__global__ __launch_bounds__(256)
void softmax_kernel(float* __restrict__ scores, _Float16* __restrict__ attn,
                    const unsigned char* __restrict__ mask, int S, int Hn) {
  __shared__ float sdata[256];
  int i = blockIdx.x, bz = blockIdx.y, tid = threadIdx.x;
  int b = bz / Hn;
  long long roff = ((long long)bz * S + i) * S;
  float* row = scores + roff;
  _Float16* arow = attn + roff;
  const unsigned char* mrow = mask + (long long)b * S;
  float mx = -1e30f;
  for (int j = tid; j < S; j += 256) {
    float v = row[j];
    if (mrow[j]) v = -1e18f;
    row[j] = v;
    mx = fmaxf(mx, v);
  }
  sdata[tid] = mx; __syncthreads();
  for (int st = 128; st > 0; st >>= 1) { if (tid < st) sdata[tid] = fmaxf(sdata[tid], sdata[tid + st]); __syncthreads(); }
  mx = sdata[0]; __syncthreads();
  float sum = 0.f;
  for (int j = tid; j < S; j += 256) sum += __expf(row[j] - mx);
  sdata[tid] = sum; __syncthreads();
  for (int st = 128; st > 0; st >>= 1) { if (tid < st) sdata[tid] += sdata[tid + st]; __syncthreads(); }
  float inv = 1.f / sdata[0];
  for (int j = tid; j < S; j += 256) arow[j] = (_Float16)(__expf(row[j] - mx) * inv);
}

// ------------------------------ host launcher ------------------------------

static void launch_gemm(hipStream_t stream,
                        const _Float16* A, long long sA1, long long sA2, int bdivA, int lda,
                        const _Float16* B, long long sB1, long long sB2, int bdivB, int ldb, int bTrans,
                        const float* bias, const float* resid, int ldr,
                        float* Cf, _Float16* Ch,
                        long long sC1, long long sC2, int bdivC, int ldc,
                        int M, int Nmat, int K, int batch, float alpha, int relu) {
  dim3 grid(Nmat / BN, M / BM, batch);
  gemm_wmma_f16<<<grid, 256, 0, stream>>>(A, sA1, sA2, bdivA, lda,
                                          B, sB1, sB2, bdivB, ldb, bTrans,
                                          bias, resid, ldr, Cf, Ch,
                                          sC1, sC2, bdivC, ldc, M, Nmat, K, alpha, relu);
}

static void launch_cvt(hipStream_t stream, const float* in, _Float16* out, long long n) {
  f32to16_kernel<<<(int)((n + 255) / 256), 256, 0, stream>>>(in, out, n);
}

extern "C" void kernel_launch(void* const* d_in, const int* in_sizes, int n_in,
                              void* d_out, int out_size, void* d_ws, size_t ws_size,
                              hipStream_t stream) {
  const int Bv = 8, S = 512, D = 512, Hh = 8, DFF = 2048, E = 32768;
  const int N = Bv * S;                        // 4096
  const long long ND  = (long long)N * D;      // 2,097,152
  const long long NH4 = (long long)N * 4096;   // 16,777,216

  const float* inputs  = (const float*)d_in[0];
  const int*   ei      = (const int*)d_in[1];
  const unsigned char* mask = (const unsigned char*)d_in[2];
  const float* ln1_g   = (const float*)d_in[3];
  const float* ln1_b   = (const float*)d_in[4];
  const float* lin_w   = (const float*)d_in[5];
  const float* lin_b   = (const float*)d_in[6];
  const float* gat_w   = (const float*)d_in[7];
  const float* att_src = (const float*)d_in[8];
  const float* att_dst = (const float*)d_in[9];
  const float* gat_bias= (const float*)d_in[10];
  const float* ln2_g   = (const float*)d_in[11];
  const float* ln2_b   = (const float*)d_in[12];
  const float* wq = (const float*)d_in[13]; const float* bq = (const float*)d_in[14];
  const float* wk = (const float*)d_in[15]; const float* bk = (const float*)d_in[16];
  const float* wv = (const float*)d_in[17]; const float* bv = (const float*)d_in[18];
  const float* wo = (const float*)d_in[19]; const float* bo = (const float*)d_in[20];
  const float* fln_g = (const float*)d_in[21];
  const float* fln_b = (const float*)d_in[22];
  const float* w1 = (const float*)d_in[23]; const float* b1 = (const float*)d_in[24];
  const float* w2 = (const float*)d_in[25]; const float* b2 = (const float*)d_in[26];
  float* outp = (float*)d_out;

  // --- workspace (halves/floats), lifetime-based reuse, ~180 MB peak ---
  _Float16* lin16 = (_Float16*)d_ws;            // 512*512
  _Float16* gat16 = lin16 + 262144;             // 4096*512
  _Float16* wq16  = gat16 + 2097152;
  _Float16* wk16  = wq16  + 262144;
  _Float16* wv16  = wk16  + 262144;
  _Float16* wo16  = wv16  + 262144;
  _Float16* w116  = wo16  + 262144;             // 2048*512
  _Float16* w216  = w116  + 1048576;            // 512*2048
  _Float16* xln16 = w216  + 1048576;            // [N,D]  (reused: z16)
  _Float16* x116  = xln16 + ND;                 // [N,D]  (reused: y16, ctx16)
  _Float16* h16   = x116  + ND;                 // [N,4096] (reused: attn16)
  float*    agg   = (float*)(h16 + NH4);        // [N,4096] f32; sub-reused below:
  _Float16* q16   = (_Float16*)agg;             //   q/k/v  [N,D] f16 each
  _Float16* k16   = q16 + ND;
  _Float16* v16   = k16 + ND;
  float*    outb  = (float*)(v16 + ND);         //   out    [N,D] f32
  _Float16* z116  = (_Float16*)(outb + ND);     //   z1     [N,DFF] f16
  float*    scores= agg + NH4;                  // [B*H,S,S] f32 (reused: gnn)
  float*    gnn   = scores;                     //   [N,D] f32, dead before scores
  float*    asrc  = scores + NH4;
  float*    adst  = asrc + (long long)N * Hh;
  float*    mbuf  = adst + (long long)N * Hh;
  float*    dbuf  = mbuf + (long long)N * Hh;
  _Float16 *z16 = xln16, *y16 = x116, *ctx16 = x116, *attn16 = h16;

  // 0. one-time (per call) weight conversion f32 -> f16
  launch_cvt(stream, lin_w, lin16, 262144);
  launch_cvt(stream, gat_w, gat16, 2097152);
  launch_cvt(stream, wq, wq16, 262144);
  launch_cvt(stream, wk, wk16, 262144);
  launch_cvt(stream, wv, wv16, 262144);
  launch_cvt(stream, wo, wo16, 262144);
  launch_cvt(stream, w1, w116, 1048576);
  launch_cvt(stream, w2, w216, 1048576);

  // 1. LN1 -> f16
  ln_kernel<<<N, 256, 0, stream>>>(inputs, ln1_g, ln1_b, xln16, D);
  // 2. x1 = xln @ lin_w^T + lin_b        (f16 out)
  launch_gemm(stream, xln16,0,0,1,D, lin16,0,0,1,D,1, lin_b, nullptr,0,
              nullptr, x116, 0,0,1,D, N,D,D, 1, 1.f, 0);
  // 3. h = x1 @ gat_w^T                  (f16 out)
  launch_gemm(stream, x116,0,0,1,D, gat16,0,0,1,D,1, nullptr, nullptr,0,
              nullptr, h16, 0,0,1,4096, N,4096,D, 1, 1.f, 0);
  // 4. attention logits
  gat_att_kernel<<<(N * Hh) / 8, 256, 0, stream>>>(h16, att_src, att_dst, asrc, adst);
  // 5. segment softmax + aggregate (atomics)
  fill_kernel<<<(N * Hh + 255) / 256, 256, 0, stream>>>(mbuf, -1e30f, (long long)N * Hh);
  fill_kernel<<<(N * Hh + 255) / 256, 256, 0, stream>>>(dbuf, 0.f, (long long)N * Hh);
  fill_kernel<<<(int)((NH4 + 255) / 256), 256, 0, stream>>>(agg, 0.f, NH4);
  int Etot = E + N;
  edge_max_kernel<<<(Etot * 8 + 255) / 256, 256, 0, stream>>>(ei, asrc, adst, mbuf, E, N);
  edge_sum_kernel<<<(Etot * 8 + 255) / 256, 256, 0, stream>>>(ei, asrc, adst, mbuf, dbuf, E, N);
  edge_agg_kernel<<<Etot, 256, 0, stream>>>(ei, asrc, adst, mbuf, dbuf, h16, agg, E, N);
  // 6. head mean + bias -> gnn (f32)
  head_mean_kernel<<<(int)((ND + 255) / 256), 256, 0, stream>>>(agg, gat_bias, gnn, ND);
  // 7. LN2 -> y (f16)
  ln_kernel<<<N, 256, 0, stream>>>(gnn, ln2_g, ln2_b, y16, D);
  // 8. q/k/v (f16 out; q scaled by 1/sqrt(64))
  launch_gemm(stream, y16,0,0,1,D, wq16,0,0,1,D,1, bq, nullptr,0, nullptr, q16, 0,0,1,D, N,D,D, 1, 0.125f, 0);
  launch_gemm(stream, y16,0,0,1,D, wk16,0,0,1,D,1, bk, nullptr,0, nullptr, k16, 0,0,1,D, N,D,D, 1, 1.f, 0);
  launch_gemm(stream, y16,0,0,1,D, wv16,0,0,1,D,1, bv, nullptr,0, nullptr, v16, 0,0,1,D, N,D,D, 1, 1.f, 0);
  // 9. scores[b,h] = q_bh @ k_bh^T  (batch 64, f32 out)
  launch_gemm(stream, q16,(long long)S*D,64,Hh,D, k16,(long long)S*D,64,Hh,D,1,
              nullptr, nullptr,0, scores, nullptr, (long long)S*S,0,1,S, S,S,64, Bv*Hh, 1.f, 0);
  // 10. softmax rows (mask) -> f16 attn
  { dim3 g(S, Bv * Hh); softmax_kernel<<<g, 256, 0, stream>>>(scores, attn16, mask, S, Hh); }
  // 11. ctx[b,h] = attn @ v_bh  (B non-transposed, f16 out into [B,S,H*dh])
  launch_gemm(stream, attn16,(long long)S*S,0,1,S, v16,(long long)S*D,64,Hh,D,0,
              nullptr, nullptr,0, nullptr, ctx16, (long long)S*D,64,Hh,D, S,64,S, Bv*Hh, 1.f, 0);
  // 12. out = ctx @ wo^T + bo + inputs  (f32 out)
  launch_gemm(stream, ctx16,0,0,1,D, wo16,0,0,1,D,1, bo, inputs,D,
              outb, nullptr, 0,0,1,D, N,D,D, 1, 1.f, 0);
  // 13. LN3 -> z (f16)
  ln_kernel<<<N, 256, 0, stream>>>(outb, fln_g, fln_b, z16, D);
  // 14. z1 = relu(z @ w1^T + b1)  (f16 out)
  launch_gemm(stream, z16,0,0,1,D, w116,0,0,1,D,1, b1, nullptr,0,
              nullptr, z116, 0,0,1,DFF, N,DFF,D, 1, 1.f, 1);
  // 15. d_out = z1 @ w2^T + b2 + outb  (f32 out)
  launch_gemm(stream, z116,0,0,1,DFF, w216,0,0,1,DFF,1, b2, outb,D,
              outp, nullptr, 0,0,1,D, N,D,DFF, 1, 1.f, 0);
}
